// dMag0Grid_58566174048365
// MI455X (gfx1250) — compile-verified, matrix operand order checked by reading, not run on hardware
//
#include <hip/hip_runtime.h>
#include <stdint.h>

typedef __attribute__((ext_vector_type(16))) _Float16 v16h;
typedef __attribute__((ext_vector_type(8)))  float    v8f;

#define NTINT 30   // n_tint (reference: 30)
#define NA    200  // n_alpha (reference: 200)
#define TPB   256  // 8 waves of 32

__global__ __launch_bounds__(TPB) void pdet_kernel(
    const float* __restrict__ alpha,      // (norb, ntimes)
    const float* __restrict__ dMag,       // (norb, ntimes)
    const float* __restrict__ fZ_vals,    // (ntimes,)
    const float* __restrict__ kEZ_val,    // scalar
    const float* __restrict__ grid,       // (n_fZ, n_kEZ, NTINT, NA)
    const float* __restrict__ kEZs,       // (n_kEZ,)
    const float* __restrict__ alphas,     // (NA,)
    const float* __restrict__ log_fZs,    // (n_fZ,)
    const float* __restrict__ log_alphas, // (NA,)
    float* __restrict__ out,              // (ntimes, NTINT)
    int norb, int ntimes, int n_fZ, int n_kEZ, int n_alpha)
{
    __shared__ float    gsub[NTINT * NA];   // 24000 B: grid[fZ0, kEZ_ind, :, :]
    __shared__ _Float16 cnt[32 * TPB];      // 16384 B: [j][thread] f16 counts

    const int tid = threadIdx.x;
    const int t   = blockIdx.x;

    // ---- per-block scalar setup (redundant per thread; cheap) ----
    const float lf0    = log_fZs[0];
    const float inv_lf = 1.0f / (log_fZs[1] - lf0);
    float fZ_ind = (log10f(fZ_vals[t]) - lf0) * inv_lf;
    int fZ0 = (int)floorf(fZ_ind) + 1;
    fZ0 = fZ0 < 0 ? 0 : (fZ0 > n_fZ - 2 ? n_fZ - 2 : fZ0);

    const float kv = kEZ_val[0];
    int kEZ_ind = -1;                      // searchsorted(right) - 1
    for (int i = 0; i < n_kEZ; ++i) kEZ_ind += (kEZs[i] <= kv) ? 1 : 0;
    if (kEZ_ind < 0) kEZ_ind += n_kEZ;     // python-style negative wrap

    // ---- async copy grid row (24 KB, contiguous, 16B aligned) into LDS ----
    const float* gptr = grid + (size_t)(fZ0 * n_kEZ + kEZ_ind) * (size_t)(NTINT * NA);
    const unsigned lds_base = (unsigned)(size_t)(&gsub[0]);
    for (int i = tid; i < (NTINT * NA) / 4; i += TPB) {
        unsigned laddr = lds_base + (unsigned)i * 16u;
        uint64_t gaddr = (uint64_t)(gptr + i * 4);
        asm volatile("global_load_async_to_lds_b128 %0, %1, off"
                     :: "v"(laddr), "v"(gaddr) : "memory");
    }
    asm volatile("s_wait_asynccnt 0x0" ::: "memory");
    __syncthreads();

    // ---- per-thread orbit loop ----
    const float la0    = log_alphas[0];
    const float inv_la = 1.0f / (log_alphas[1] - la0);
    const float alo    = alphas[0];
    const float ahi    = alphas[n_alpha - 1];

    unsigned lc[NTINT];
    #pragma unroll
    for (int j = 0; j < NTINT; ++j) lc[j] = 0u;

    for (int o = tid; o < norb; o += TPB) {
        const float a  = alpha[(size_t)o * (size_t)ntimes + t];
        const float dm = dMag [(size_t)o * (size_t)ntimes + t];
        const bool geom = (a >= alo) && (a <= ahi);
        const float a_ind = (log10f(a) - la0) * inv_la;
        int a0 = (int)a_ind;                              // trunc toward zero == astype(int32)
        a0 = a0 < 0 ? 0 : (a0 > n_alpha - 1 ? n_alpha - 1 : a0);
        const float dal = a_ind - (float)a0;
        const int a0s = a0 < n_alpha - 2 ? a0 : n_alpha - 2;
        if (geom) {
            #pragma unroll
            for (int j = 0; j < NTINT; ++j) {
                const float g0 = gsub[j * NA + a0s];
                const float g1 = gsub[j * NA + a0s + 1];
                const float dim = g0 + dal * (g1 - g0);
                lc[j] += (dm - dim < 0.0f) ? 1u : 0u;     // signbit(dMag - dim)
            }
        }
    }

    // ---- stage counts to LDS as f16 in [j][thread] layout (rows 30,31 zero) ----
    #pragma unroll
    for (int j = 0; j < 32; ++j) {
        float v = (j < NTINT) ? (float)lc[j] : 0.0f;      // counts <= 8: exact in f16
        cnt[j * TPB + tid] = (_Float16)v;
    }
    __syncthreads();

    // ---- WMMA column-sum: ones(16x32) x B(32x16), 8 K-chunks cover 256 threads ----
    if (tid < 32) {                                       // wave 0, EXEC all ones
        const int lane = tid;
        const int n16  = lane & 15;                       // B: N = lane % 16
        const int hiK  = (lane >> 4) & 1;                 // high half-lanes take upper 16 K rows

        v16h aONE;
        #pragma unroll
        for (int i = 0; i < 16; ++i) aONE[i] = (_Float16)1.0f;

        const unsigned* cnt32 = (const unsigned*)cnt;     // packed f16 pairs

        #pragma unroll
        for (int nt = 0; nt < 2; ++nt) {                  // j tiles: 0..15, 16..29(+pad)
            const int j = nt * 16 + n16;
            v8f c = {};
            #pragma unroll
            for (int kc = 0; kc < 8; ++kc) {
                const int thrbase = kc * 32 + hiK * 16;
                union { v16h h; unsigned u[8]; } b;
                #pragma unroll
                for (int v = 0; v < 8; ++v) {
                    // pair of consecutive thread counts for row j (thr even -> dword aligned)
                    b.u[v] = cnt32[(j * TPB + thrbase + 2 * v) >> 1];
                }
                c = __builtin_amdgcn_wmma_f32_16x16x32_f16(
                        /*neg_a=*/false, aONE, /*neg_b=*/false, b.h,
                        /*c_mod=*/(short)0, c, /*reuse_a=*/false, /*reuse_b=*/false);
            }
            // D row M=0 lives in c[0], lanes 0..15 hold N=0..15
            if (lane < 16 && j < NTINT) {
                out[(size_t)t * NTINT + j] = c[0] / (float)norb;   // mean over orbits
            }
        }
    }
}

extern "C" void kernel_launch(void* const* d_in, const int* in_sizes, int n_in,
                              void* d_out, int out_size, void* d_ws, size_t ws_size,
                              hipStream_t stream) {
    const float* alpha      = (const float*)d_in[0];
    const float* dMag       = (const float*)d_in[1];
    const float* fZ_vals    = (const float*)d_in[2];
    const float* kEZ_val    = (const float*)d_in[3];
    const float* grid       = (const float*)d_in[4];
    const float* kEZs       = (const float*)d_in[5];
    const float* alphas     = (const float*)d_in[6];
    const float* log_fZs    = (const float*)d_in[7];
    const float* log_alphas = (const float*)d_in[8];

    const int ntimes  = in_sizes[2];
    const int norb    = in_sizes[0] / ntimes;
    const int n_kEZ   = in_sizes[5];
    const int n_fZ    = in_sizes[7];
    const int n_alpha = in_sizes[8];

    pdet_kernel<<<ntimes, TPB, 0, stream>>>(
        alpha, dMag, fZ_vals, kEZ_val, grid, kEZs, alphas, log_fZs, log_alphas,
        (float*)d_out, norb, ntimes, n_fZ, n_kEZ, n_alpha);
}
